// TextPathLoss_4724464025998
// MI455X (gfx1250) — matrix operand order repaired and checked by reading, not compiled, not run on hardware
//
#include <hip/hip_runtime.h>
#include <hip/hip_bf16.h>
#include <math.h>

#define BB 16
#define NN 1024
#define DD 256
#define NITERS 50
#define EPSV 0.1f
#define KCH 128          // K-chunk staged in LDS for the big GEMMs
#define APAD 136         // 128 + 8 bf16 pad -> 272B row stride (conflict-free)

typedef __attribute__((ext_vector_type(16))) __bf16 v16bf;
typedef __attribute__((ext_vector_type(8)))  __bf16 v8bf;
typedef __attribute__((ext_vector_type(8)))  float  v8f;

// ---------------------------------------------------------------------------
// gfx1250 async copy helpers (ASYNCcnt-tracked global->LDS DMA)
// ---------------------------------------------------------------------------
__device__ __forceinline__ uint32_t lds_addr32(const void* p) {
    // generic LDS pointer: low 32 bits are the LDS byte offset
    return (uint32_t)(uintptr_t)p;
}
__device__ __forceinline__ void async_ld_b128(uint32_t lds_off, const __bf16* g) {
    asm volatile("global_load_async_to_lds_b128 %0, %1, off"
                 :: "v"(lds_off), "v"(g)
                 : "memory");
}
__device__ __forceinline__ void wait_async0() {
    asm volatile("s_wait_asynccnt 0x0" ::: "memory");
}

// ---------------------------------------------------------------------------
// K0: zero the atomic accumulator (Lbary)
// ---------------------------------------------------------------------------
__global__ void k0_zero(float* __restrict__ p, int n) {
    int i = blockIdx.x * 256 + threadIdx.x;
    if (i < n) p[i] = 0.f;
}

// ---------------------------------------------------------------------------
// K1: per-row L2 normalize -> bf16; also write raw transposed bf16 copies
// grid (NN, BB, 2), block 256 (= D)
// ---------------------------------------------------------------------------
__global__ __launch_bounds__(256)
void k1_norm(const float* __restrict__ X, const float* __restrict__ Y,
             __bf16* __restrict__ XnH, __bf16* __restrict__ YnH,
             __bf16* __restrict__ XtH, __bf16* __restrict__ YtH) {
    int i = blockIdx.x;
    int b = blockIdx.y;
    int which = blockIdx.z;
    const float* src = which ? Y : X;
    __bf16* nh = which ? YnH : XnH;
    __bf16* th = which ? YtH : XtH;
    int d = threadIdx.x;

    float x = src[((size_t)b * NN + i) * DD + d];

    __shared__ float red[256];
    red[d] = x * x;
    __syncthreads();
    for (int s = 128; s > 0; s >>= 1) {
        if (d < s) red[d] += red[d + s];
        __syncthreads();
    }
    float nrm = sqrtf(red[0]);
    float xn = x / fmaxf(nrm, 1e-12f);

    nh[((size_t)b * NN + i) * DD + d] = (__bf16)xn;
    th[((size_t)b * DD + d) * NN + i] = (__bf16)x;
}

// ---------------------------------------------------------------------------
// K2: S = Xn · Yn^T (v_wmma_f32_16x16x32_bf16) fused into
//     logK = -100*relu(1-S) - 0.2*|i-j|/1023
// One wave per 16x16 tile; K=256 fully register-pipelined.
// grid (8, 64, BB), block 256
// ---------------------------------------------------------------------------
__global__ __launch_bounds__(256)
void k2_logk(const __bf16* __restrict__ XnH, const __bf16* __restrict__ YnH,
             float* __restrict__ logK) {
    int wave = threadIdx.x >> 5;
    int lane = threadIdx.x & 31;
    int nt = blockIdx.x * 8 + wave;
    int mt = blockIdx.y;
    int b  = blockIdx.z;
    int m0 = mt * 16, n0 = nt * 16;
    int half = lane >> 4;
    int l15  = lane & 15;

    const __bf16* Abase = XnH + ((size_t)b * NN + (m0 + l15)) * DD;
    const __bf16* Bbase = YnH + ((size_t)b * NN + (n0 + l15)) * DD;

    v8f acc = {};
#pragma unroll
    for (int kk = 0; kk < DD; kk += 32) {
        union { v16bf v; v8bf h[2]; } au, bu;
        au.h[0] = *(const v8bf*)(Abase + kk + 8 * half);
        au.h[1] = *(const v8bf*)(Abase + kk + 16 + 8 * half);
        const __bf16* bp = Bbase + kk;
        bu.h[0] = *(const v8bf*)(bp + 8 * half);
        bu.h[1] = *(const v8bf*)(bp + 16 + 8 * half);
        acc = __builtin_amdgcn_wmma_f32_16x16x32_bf16(
            false, au.v, false, bu.v, (short)0, acc, false, false);
    }

    int n = n0 + l15;
#pragma unroll
    for (int vr = 0; vr < 8; ++vr) {
        int m = m0 + half * 8 + vr;
        float s = acc[vr];
        float lk = -100.f * fmaxf(1.f - s, 0.f)
                   - 0.2f * fabsf((float)(m - n)) * (1.f / 1023.f);
        logK[((size_t)b * NN + m) * NN + n] = lk;
    }
}

// ---------------------------------------------------------------------------
// K3: 50 log-domain Sinkhorn iterations, one batch per workgroup.
// grid BB, block 1024 (32 waves); logK stays L2-resident (64MB < 192MB L2)
// ---------------------------------------------------------------------------
__global__ __launch_bounds__(1024)
void k3_sinkhorn(const float* __restrict__ logK,
                 float* __restrict__ rOut, float* __restrict__ cOut,
                 float* __restrict__ lmainRow,
                 __bf16* __restrict__ TH, __bf16* __restrict__ TtH) {
    int b = blockIdx.x;
    int tid = threadIdx.x;
    int wave = tid >> 5, lane = tid & 31;
    __shared__ float fs[NN];
    __shared__ float gs[NN];
    const float loga = __logf(1.0f / NN + 1e-12f);

    for (int i = tid; i < NN; i += 1024) { fs[i] = 0.f; gs[i] = 0.f; }
    __syncthreads();

    const float* LK = logK + (size_t)b * NN * NN;

    for (int it = 0; it < NITERS; ++it) {
        // f[i] = loga - LSE_j( logK[i][j] + g[j] )   (wave per row)
        for (int rr = 0; rr < 32; ++rr) {
            int i = wave * 32 + rr;
            const float* row = LK + (size_t)i * NN;
            float m = -INFINITY, s = 0.f;
            for (int j = lane; j < NN; j += 32) {
                __builtin_prefetch(row + j + 256, 0, 0);
                float x = row[j] + gs[j];
                float M = fmaxf(m, x);
                s = s * __expf(m - M) + __expf(x - M);
                m = M;
            }
#pragma unroll
            for (int off = 16; off > 0; off >>= 1) {
                float m2 = __shfl_xor(m, off, 32);
                float s2 = __shfl_xor(s, off, 32);
                float M = fmaxf(m, m2);
                s = s * __expf(m - M) + s2 * __expf(m2 - M);
                m = M;
            }
            if (lane == 0) fs[i] = loga - (m + __logf(s));
        }
        __syncthreads();
        // g[j] = loga - LSE_i( logK[i][j] + f[i] )   (lane per column)
        {
            int j = wave * 32 + lane;
            float m = -INFINITY, s = 0.f;
            for (int i = 0; i < NN; ++i) {
                float x = LK[(size_t)i * NN + j] + fs[i];
                float M = fmaxf(m, x);
                s = s * __expf(m - M) + __expf(x - M);
                m = M;
            }
            gs[j] = loga - (m + __logf(s));
        }
        __syncthreads();
    }

    // finalize rows: T, T^T, r, per-row L_main
    for (int rr = 0; rr < 32; ++rr) {
        int i = wave * 32 + rr;
        const float* row = LK + (size_t)i * NN;
        float fi = fs[i];
        float rs = 0.f, lm = 0.f;
        for (int j = lane; j < NN; j += 32) {
            float lk = row[j];
            float t = __expf(lk + fi + gs[j]);
            rs += t;
            float cc = -EPSV * lk - 0.02f * fabsf((float)(i - j)) * (1.f / 1023.f);
            lm += t * cc;
            TH [((size_t)b * NN + i) * NN + j] = (__bf16)t;
            TtH[((size_t)b * NN + j) * NN + i] = (__bf16)t;
        }
#pragma unroll
        for (int off = 16; off > 0; off >>= 1) {
            rs += __shfl_xor(rs, off, 32);
            lm += __shfl_xor(lm, off, 32);
        }
        if (lane == 0) {
            rOut[(size_t)b * NN + i] = rs;
            lmainRow[(size_t)b * NN + i] = lm;
        }
    }
    // finalize columns: c
    {
        int j = wave * 32 + lane;
        float gj = gs[j];
        float cs = 0.f;
        for (int i = 0; i < NN; ++i)
            cs += __expf(LK[(size_t)i * NN + j] + fs[i] + gj);
        cOut[(size_t)b * NN + j] = cs;
    }
}

// ---------------------------------------------------------------------------
// K5: Y_bar = T @ Y. All 8 waves share the same 16-row A strip of T:
//     stage it in LDS via global_load_async_to_lds_b128 (double buffered),
//     B (Y^T rows) streams straight from global. Then accumulate
//     mean((X - Ybar/r)^2) into Lbary[b].
// grid (2, 64, BB), block 256
// ---------------------------------------------------------------------------
__global__ __launch_bounds__(256)
void k5_ybar(const __bf16* __restrict__ TH, const __bf16* __restrict__ YtH,
             const float* __restrict__ X, const float* __restrict__ rArr,
             float* __restrict__ Lbary) {
    __shared__ __align__(16) __bf16 Abuf[2][16 * APAD];
    int tid = threadIdx.x;
    int wave = tid >> 5, lane = tid & 31;
    int nt = blockIdx.x * 8 + wave;   // 0..15 (D tiles)
    int mt = blockIdx.y;              // 0..63 (N tiles)
    int b  = blockIdx.z;
    int m0 = mt * 16, n0 = nt * 16;
    int half = lane >> 4, l15 = lane & 31 & 15;

    // async copy map: 256 threads x 16B = one 16x128 bf16 chunk (4KB)
    int arow = tid >> 4;              // 0..15
    int aseg = tid & 15;              // 0..15
    const __bf16* Ag = TH + ((size_t)b * NN + (m0 + arow)) * NN + aseg * 8;
    uint32_t aoff[2];
    aoff[0] = lds_addr32(&Abuf[0][arow * APAD + aseg * 8]);
    aoff[1] = lds_addr32(&Abuf[1][arow * APAD + aseg * 8]);

    const __bf16* Bbase = YtH + ((size_t)b * DD + (n0 + l15)) * NN;

    async_ld_b128(aoff[0], Ag);
    wait_async0();
    __syncthreads();

    v8f acc = {};
    const int NCHK = NN / KCH;        // 8
    for (int c = 0; c < NCHK; ++c) {
        if (c + 1 < NCHK)
            async_ld_b128(aoff[(c + 1) & 1], Ag + (size_t)(c + 1) * KCH);
        const __bf16* Ach = &Abuf[c & 1][0];
        const __bf16* Bc  = Bbase + c * KCH;
#pragma unroll
        for (int kk = 0; kk < KCH; kk += 32) {
            union { v16bf v; v8bf h[2]; } au, bu;
            const __bf16* ap = Ach + l15 * APAD + kk;
            au.h[0] = *(const v8bf*)(ap + 8 * half);
            au.h[1] = *(const v8bf*)(ap + 16 + 8 * half);
            const __bf16* bp = Bc + kk;
            bu.h[0] = *(const v8bf*)(bp + 8 * half);
            bu.h[1] = *(const v8bf*)(bp + 16 + 8 * half);
            acc = __builtin_amdgcn_wmma_f32_16x16x32_bf16(
                false, au.v, false, bu.v, (short)0, acc, false, false);
        }
        wait_async0();
        __syncthreads();
    }

    float part = 0.f;
    int n = n0 + l15;
#pragma unroll
    for (int vr = 0; vr < 8; ++vr) {
        int m = m0 + half * 8 + vr;
        float yb = acc[vr] / (rArr[(size_t)b * NN + m] + 1e-8f);
        float diff = X[((size_t)b * NN + m) * DD + n] - yb;
        part += diff * diff;
    }
#pragma unroll
    for (int off = 16; off > 0; off >>= 1) part += __shfl_xor(part, off, 32);
    if (lane == 0) atomicAdd(&Lbary[b], part * (1.0f / ((float)NN * (float)DD)));
}

// ---------------------------------------------------------------------------
// K6: X_bar = T^T @ X, same LDS-staged structure (A = T^T, B = X^T),
//     accumulate mean((Y - Xbar/c)^2) into Lbary[b].
// ---------------------------------------------------------------------------
__global__ __launch_bounds__(256)
void k6_xbar(const __bf16* __restrict__ TtH, const __bf16* __restrict__ XtH,
             const float* __restrict__ Y, const float* __restrict__ cArr,
             float* __restrict__ Lbary) {
    __shared__ __align__(16) __bf16 Abuf[2][16 * APAD];
    int tid = threadIdx.x;
    int wave = tid >> 5, lane = tid & 31;
    int nt = blockIdx.x * 8 + wave;
    int mt = blockIdx.y;
    int b  = blockIdx.z;
    int m0 = mt * 16, n0 = nt * 16;
    int half = lane >> 4, l15 = lane & 15;

    int arow = tid >> 4;
    int aseg = tid & 15;
    const __bf16* Ag = TtH + ((size_t)b * NN + (m0 + arow)) * NN + aseg * 8;
    uint32_t aoff[2];
    aoff[0] = lds_addr32(&Abuf[0][arow * APAD + aseg * 8]);
    aoff[1] = lds_addr32(&Abuf[1][arow * APAD + aseg * 8]);

    const __bf16* Bbase = XtH + ((size_t)b * DD + (n0 + l15)) * NN;

    async_ld_b128(aoff[0], Ag);
    wait_async0();
    __syncthreads();

    v8f acc = {};
    const int NCHK = NN / KCH;
    for (int c = 0; c < NCHK; ++c) {
        if (c + 1 < NCHK)
            async_ld_b128(aoff[(c + 1) & 1], Ag + (size_t)(c + 1) * KCH);
        const __bf16* Ach = &Abuf[c & 1][0];
        const __bf16* Bc  = Bbase + c * KCH;
#pragma unroll
        for (int kk = 0; kk < KCH; kk += 32) {
            union { v16bf v; v8bf h[2]; } au, bu;
            const __bf16* ap = Ach + l15 * APAD + kk;
            au.h[0] = *(const v8bf*)(ap + 8 * half);
            au.h[1] = *(const v8bf*)(ap + 16 + 8 * half);
            const __bf16* bp = Bc + kk;
            bu.h[0] = *(const v8bf*)(bp + 8 * half);
            bu.h[1] = *(const v8bf*)(bp + 16 + 8 * half);
            acc = __builtin_amdgcn_wmma_f32_16x16x32_bf16(
                false, au.v, false, bu.v, (short)0, acc, false, false);
        }
        wait_async0();
        __syncthreads();
    }

    float part = 0.f;
    int n = n0 + l15;
#pragma unroll
    for (int vr = 0; vr < 8; ++vr) {
        int m = m0 + half * 8 + vr;
        float xb = acc[vr] / (cArr[(size_t)b * NN + m] + 1e-8f);
        float diff = Y[((size_t)b * NN + m) * DD + n] - xb;
        part += diff * diff;
    }
#pragma unroll
    for (int off = 16; off > 0; off >>= 1) part += __shfl_xor(part, off, 32);
    if (lane == 0) atomicAdd(&Lbary[b], part * (1.0f / ((float)NN * (float)DD)));
}

// ---------------------------------------------------------------------------
// K7: per-batch reductions: L_main, sum(r), sum(c), x_g, y_g, L_cos_global
// grid BB, block 256
// ---------------------------------------------------------------------------
__global__ __launch_bounds__(256)
void k7_stats(const float* __restrict__ X, const float* __restrict__ Y,
              const float* __restrict__ rArr, const float* __restrict__ cArr,
              const float* __restrict__ lmainRow,
              float* __restrict__ LmainOut, float* __restrict__ LcosOut) {
    int b = blockIdx.x;
    int t = threadIdx.x;
    __shared__ float rs[NN];
    __shared__ float cs[NN];
    __shared__ float red[256];

    for (int i = t; i < NN; i += 256) {
        rs[i] = rArr[(size_t)b * NN + i];
        cs[i] = cArr[(size_t)b * NN + i];
    }
    __syncthreads();

    float lm = 0.f, sr = 0.f, sc = 0.f;
    for (int i = t; i < NN; i += 256) {
        lm += lmainRow[(size_t)b * NN + i];
        sr += rs[i];
        sc += cs[i];
    }

    auto blockReduce = [&](float v) -> float {
        red[t] = v;
        __syncthreads();
        for (int s2 = 128; s2 > 0; s2 >>= 1) {
            if (t < s2) red[t] += red[t + s2];
            __syncthreads();
        }
        float r0 = red[0];
        __syncthreads();
        return r0;
    };

    float Lmain = blockReduce(lm);
    float sumr  = blockReduce(sr);
    float sumc  = blockReduce(sc);

    float xg = 0.f, yg = 0.f;
    for (int i = 0; i < NN; ++i) {
        xg += X[((size_t)b * NN + i) * DD + t] * rs[i];
        yg += Y[((size_t)b * NN + i) * DD + t] * cs[i];
    }
    xg /= (sumr + 1e-8f);
    yg /= (sumc + 1e-8f);

    float nx = blockReduce(xg * xg);
    float ny = blockReduce(yg * yg);
    float dt = blockReduce(xg * yg);

    if (t == 0) {
        float denom = fmaxf(sqrtf(nx), 1e-12f) * fmaxf(sqrtf(ny), 1e-12f);
        LcosOut[b] = 1.f - dt / denom;
        LmainOut[b] = Lmain;
    }
}

// ---------------------------------------------------------------------------
// K8: final mean over batch
// ---------------------------------------------------------------------------
__global__ void k8_final(const float* __restrict__ Lmain,
                         const float* __restrict__ Lbary,
                         const float* __restrict__ Lcos,
                         float* __restrict__ out) {
    int lane = threadIdx.x;
    float v = 0.f;
    if (lane < BB) v = Lmain[lane] + 0.5f * Lbary[lane] + 0.2f * Lcos[lane];
#pragma unroll
    for (int off = 16; off > 0; off >>= 1) v += __shfl_xor(v, off, 32);
    if (lane == 0) out[0] = v / (float)BB;
}

// ---------------------------------------------------------------------------
extern "C" void kernel_launch(void* const* d_in, const int* in_sizes, int n_in,
                              void* d_out, int out_size, void* d_ws, size_t ws_size,
                              hipStream_t stream) {
    const float* X = (const float*)d_in[0];
    const float* Y = (const float*)d_in[1];

    char* w = (char*)d_ws;
    const size_t nBD  = (size_t)BB * NN * DD;
    const size_t nBNN = (size_t)BB * NN * NN;

    __bf16* XnH = (__bf16*)w; w += nBD * 2;
    __bf16* YnH = (__bf16*)w; w += nBD * 2;
    __bf16* XtH = (__bf16*)w; w += nBD * 2;
    __bf16* YtH = (__bf16*)w; w += nBD * 2;
    float*  logK = (float*)w; w += nBNN * 4;
    __bf16* TH   = (__bf16*)w; w += nBNN * 2;
    __bf16* TtH  = (__bf16*)w; w += nBNN * 2;
    float*  rArr = (float*)w; w += (size_t)BB * NN * 4;
    float*  cArr = (float*)w; w += (size_t)BB * NN * 4;
    float*  lmr  = (float*)w; w += (size_t)BB * NN * 4;
    float*  Lmain = (float*)w; w += BB * 4;
    float*  Lbary = (float*)w; w += BB * 4;
    float*  Lcos  = (float*)w; w += BB * 4;

    k0_zero<<<1, 256, 0, stream>>>(Lbary, BB);
    k1_norm<<<dim3(NN, BB, 2), 256, 0, stream>>>(X, Y, XnH, YnH, XtH, YtH);
    k2_logk<<<dim3(8, 64, BB), 256, 0, stream>>>(XnH, YnH, logK);
    k3_sinkhorn<<<BB, 1024, 0, stream>>>(logK, rArr, cArr, lmr, TH, TtH);
    k5_ybar<<<dim3(2, 64, BB), 256, 0, stream>>>(TH, YtH, X, rArr, Lbary);
    k6_xbar<<<dim3(2, 64, BB), 256, 0, stream>>>(TtH, XtH, Y, cArr, Lbary);
    k7_stats<<<BB, 256, 0, stream>>>(X, Y, rArr, cArr, lmr, Lmain, Lcos);
    k8_final<<<1, 32, 0, stream>>>(Lmain, Lbary, Lcos, (float*)d_out);
}
